// Encoder_65884798320851
// MI455X (gfx1250) — compile-verified, matrix operand order checked by reading, not run on hardware
//
#include <hip/hip_runtime.h>
#include <hip/hip_bf16.h>
#include <math.h>

typedef __attribute__((ext_vector_type(16))) __bf16 v16bf;
typedef __attribute__((ext_vector_type(8)))  float  v8f;

__device__ __forceinline__ float sigmf(float x) { return 1.f / (1.f + expf(-x)); }

// ---------------------------------------------------------------------------
// Strict batched WMMA GEMM: Co[b] = A[b](MxK, row-major) * Bm[b](KxN, row-major)
// REQUIRES: M % 64 == 0 (grid.y = M/64), N % 16 == 0, K % 32 == 0,
//           lda/ldb/ldc % 4 == 0, base pointers 16B-aligned.
// fp32 in/out, bf16 fragments, f32 accumulation. Optional per-row bias + ReLU.
// Block = 128 threads = 4 waves stacked along M; each wave owns a 16x32 C tile
// (two accumulators) so each A fragment feeds two v_wmma.
// B tile (32k x 32n) staged via LDS, transposed, stride 36 floats
// (16B-aligned float4 reads, <=2-way bank conflicts).
// grid = ((N+31)/32, M/64, batch); N % 32 == 16 edge: second 16 columns are
// zero-filled ONCE before the K loop (block-uniform), so the hot loop is
// guard-free in all blocks.
// ---------------------------------------------------------------------------
__global__ void wmma_gemm_kernel(const float* __restrict__ A,
                                 const float* __restrict__ Bm,
                                 float* __restrict__ Co,
                                 const float* __restrict__ biasM,
                                 int M, int N, int K,
                                 int lda, int ldb, int ldc,
                                 long long sA, long long sB, long long sC,
                                 int relu)
{
    A  += (long long)blockIdx.z * sA;
    Bm += (long long)blockIdx.z * sB;
    Co += (long long)blockIdx.z * sC;

    const int wave = threadIdx.x >> 5;
    const int lane = threadIdx.x & 31;
    const int lm   = lane & 15;   // row (A) / col-within-16-tile (B)
    const int kh   = lane >> 4;   // k-half selector per ISA 16-bit fragment layout
    const int m0   = (blockIdx.y * 4 + wave) * 16;
    const int n0   = blockIdx.x * 32;
    const bool haveN1 = (n0 + 16) < N;   // second 16-wide N tile valid?

    __shared__ float sBt[32][36];  // [n][k] transposed B tile, padded stride

    // Per-thread staging coordinates: col = tid&15, rows = (tid>>4) + {0,8,16,24}
    const int sc = threadIdx.x & 15;
    const int sr = threadIdx.x >> 4;   // 0..7

    // Edge block column: zero columns 16..31 once; never overwritten below.
    if (!haveN1) {
        float* lZ = &sBt[sc + 16][sr];
#pragma unroll
        for (int q = 0; q < 4; ++q) lZ[8 * q] = 0.f;
    }

    const float* Ap = A + (long long)(m0 + lm) * lda;  // advances by 32 per step
    const float* Bp = Bm + n0;                          // advances by 32*ldb

    v8f acc0 = {0.f, 0.f, 0.f, 0.f, 0.f, 0.f, 0.f, 0.f};
    v8f acc1 = {0.f, 0.f, 0.f, 0.f, 0.f, 0.f, 0.f, 0.f};

    for (int k0 = 0; k0 < K; k0 += 32) {
        // ---- stage B tile: 32k x 32n, coalesced read, transposed LDS write ----
        {
            const float* gB = Bp + (long long)sr * ldb + sc;
            float* lB = &sBt[sc][sr];
#pragma unroll
            for (int q = 0; q < 4; ++q)
                lB[8 * q] = gB[(long long)(8 * q) * ldb];
            if (haveN1) {
                const float* gB1 = gB + 16;
                float* lB1 = &sBt[sc + 16][sr];
#pragma unroll
                for (int q = 0; q < 4; ++q)
                    lB1[8 * q] = gB1[(long long)(8 * q) * ldb];
            }
        }
        // prefetch next tiles while this one is consumed (global_prefetch_b8)
        if (k0 + 32 < K) {
            __builtin_prefetch(Ap + 32, 0, 1);
            __builtin_prefetch(Bp + (long long)32 * ldb + lane, 0, 1);
        }
        __syncthreads();

        // ---- A fragment: two contiguous 8-float runs -> 4x global b128 ----
        const float4 a0 = *(const float4*)(Ap + 8 * kh);
        const float4 a1 = *(const float4*)(Ap + 8 * kh + 4);
        const float4 a2 = *(const float4*)(Ap + 16 + 8 * kh);
        const float4 a3 = *(const float4*)(Ap + 16 + 8 * kh + 4);

        v16bf af;
        af[0]  = (__bf16)a0.x; af[1]  = (__bf16)a0.y; af[2]  = (__bf16)a0.z; af[3]  = (__bf16)a0.w;
        af[4]  = (__bf16)a1.x; af[5]  = (__bf16)a1.y; af[6]  = (__bf16)a1.z; af[7]  = (__bf16)a1.w;
        af[8]  = (__bf16)a2.x; af[9]  = (__bf16)a2.y; af[10] = (__bf16)a2.z; af[11] = (__bf16)a2.w;
        af[12] = (__bf16)a3.x; af[13] = (__bf16)a3.y; af[14] = (__bf16)a3.z; af[15] = (__bf16)a3.w;

        // ---- B fragments from LDS: columns lm and 16+lm, contiguous in k ----
        {
            const float4 b0 = *(const float4*)(&sBt[lm][8 * kh]);
            const float4 b1 = *(const float4*)(&sBt[lm][8 * kh + 4]);
            const float4 b2 = *(const float4*)(&sBt[lm][16 + 8 * kh]);
            const float4 b3 = *(const float4*)(&sBt[lm][16 + 8 * kh + 4]);
            v16bf bf;
            bf[0]  = (__bf16)b0.x; bf[1]  = (__bf16)b0.y; bf[2]  = (__bf16)b0.z; bf[3]  = (__bf16)b0.w;
            bf[4]  = (__bf16)b1.x; bf[5]  = (__bf16)b1.y; bf[6]  = (__bf16)b1.z; bf[7]  = (__bf16)b1.w;
            bf[8]  = (__bf16)b2.x; bf[9]  = (__bf16)b2.y; bf[10] = (__bf16)b2.z; bf[11] = (__bf16)b2.w;
            bf[12] = (__bf16)b3.x; bf[13] = (__bf16)b3.y; bf[14] = (__bf16)b3.z; bf[15] = (__bf16)b3.w;
            acc0 = __builtin_amdgcn_wmma_f32_16x16x32_bf16(false, af, false, bf,
                                                           (short)0, acc0, false, false);
        }
        {
            const float4 b0 = *(const float4*)(&sBt[16 + lm][8 * kh]);
            const float4 b1 = *(const float4*)(&sBt[16 + lm][8 * kh + 4]);
            const float4 b2 = *(const float4*)(&sBt[16 + lm][16 + 8 * kh]);
            const float4 b3 = *(const float4*)(&sBt[16 + lm][16 + 8 * kh + 4]);
            v16bf bf;
            bf[0]  = (__bf16)b0.x; bf[1]  = (__bf16)b0.y; bf[2]  = (__bf16)b0.z; bf[3]  = (__bf16)b0.w;
            bf[4]  = (__bf16)b1.x; bf[5]  = (__bf16)b1.y; bf[6]  = (__bf16)b1.z; bf[7]  = (__bf16)b1.w;
            bf[8]  = (__bf16)b2.x; bf[9]  = (__bf16)b2.y; bf[10] = (__bf16)b2.z; bf[11] = (__bf16)b2.w;
            bf[12] = (__bf16)b3.x; bf[13] = (__bf16)b3.y; bf[14] = (__bf16)b3.z; bf[15] = (__bf16)b3.w;
            acc1 = __builtin_amdgcn_wmma_f32_16x16x32_bf16(false, af, false, bf,
                                                           (short)0, acc1, false, false);
        }
        __syncthreads();
        Ap += 32;
        Bp += (long long)32 * ldb;
    }

    // C/D layout: lane -> n = lane%16, mhalf = lane/16; VGPR i -> m = i + 8*mhalf
#pragma unroll
    for (int i = 0; i < 8; ++i) {
        const int m = m0 + i + 8 * kh;
        float bv = biasM ? biasM[m] : 0.f;
        {
            float v = acc0[i] + bv;
            if (relu) v = fmaxf(v, 0.f);
            Co[(long long)m * ldc + (n0 + lm)] = v;
        }
        if (haveN1) {
            float v = acc1[i] + bv;
            if (relu) v = fmaxf(v, 0.f);
            Co[(long long)m * ldc + (n0 + 16 + lm)] = v;
        }
    }
}

// ---------------------------------------------------------------------------
// conv2d stride 2, 7x7, pad 3 (NCHW / OIHW)
// ---------------------------------------------------------------------------
__global__ void conv2d_s2_kernel(const float* __restrict__ x, const float* __restrict__ w,
                                 float* __restrict__ y,
                                 int B, int Cin, int Hin, int Win,
                                 int Cout, int Hout, int Wout)
{
    int t = blockIdx.x * blockDim.x + threadIdx.x;
    int total = B * Cout * Hout * Wout;
    if (t >= total) return;
    int ow = t % Wout; int r = t / Wout;
    int oh = r % Hout; r /= Hout;
    int oc = r % Cout; int b = r / Cout;
    float acc = 0.f;
    for (int ic = 0; ic < Cin; ++ic) {
        const float* xp = x + ((long long)(b * Cin + ic)) * Hin * Win;
        const float* wp = w + ((long long)(oc * Cin + ic)) * 49;
        for (int khh = 0; khh < 7; ++khh) {
            int ih = 2 * oh + khh - 3;
            if (ih < 0 || ih >= Hin) continue;
            for (int kw = 0; kw < 7; ++kw) {
                int iw = 2 * ow + kw - 3;
                if (iw < 0 || iw >= Win) continue;
                acc += xp[(long long)ih * Win + iw] * wp[khh * 7 + kw];
            }
        }
    }
    y[t] = acc;
}

// ---------------------------------------------------------------------------
// conv1d stride 2, K=7, pad 3 (NCH / OIH)
// ---------------------------------------------------------------------------
__global__ void conv1d_s2_kernel(const float* __restrict__ x, const float* __restrict__ w,
                                 float* __restrict__ y,
                                 int B, int Cin, int Lin, int Cout, int Lout)
{
    int t = blockIdx.x * blockDim.x + threadIdx.x;
    int total = B * Cout * Lout;
    if (t >= total) return;
    int ol = t % Lout; int r = t / Lout;
    int oc = r % Cout; int b = r / Cout;
    const float* wp = w + (long long)oc * Cin * 7;
    const float* xp = x + (long long)b * Cin * Lin;
    float acc = 0.f;
    for (int ic = 0; ic < Cin; ++ic)
        for (int k = 0; k < 7; ++k) {
            int il = 2 * ol + k - 3;
            if (il >= 0 && il < Lin)
                acc += xp[(long long)ic * Lin + il] * wp[ic * 7 + k];
        }
    y[t] = acc;
}

// (B,C,L) -> (B,L,C)
__global__ void transpose_CL_kernel(const float* __restrict__ x, float* __restrict__ y,
                                    int B, int C, int L)
{
    int t = blockIdx.x * blockDim.x + threadIdx.x;
    int total = B * C * L;
    if (t >= total) return;
    int l = t % L; int r = t / L;
    int c = r % C; int b = r / C;
    y[((long long)(b * L + l)) * C + c] = x[t];
}

// dt = softplus(zx[..., -H:] + dt_bias); dA = exp(dt * (-exp(A_log)))
__global__ void dtprep_kernel(const float* __restrict__ zx, const float* __restrict__ dt_bias,
                              const float* __restrict__ A_log,
                              float* __restrict__ dt, float* __restrict__ dA,
                              int BL, int H, int d_in_proj)
{
    int t = blockIdx.x * blockDim.x + threadIdx.x;
    if (t >= BL * H) return;
    int h = t % H; int bl = t / H;
    float x  = zx[(long long)bl * d_in_proj + (d_in_proj - H) + h] + dt_bias[h];
    float sp = (x > 20.f) ? x : log1pf(expf(x));
    float Ah = -expf(A_log[h]);
    dt[t] = sp;
    dA[t] = expf(sp * Ah);
}

// causal depthwise conv K=4 over xBC slice of zx, + bias, SiLU -> (B,L,Cc)
__global__ void convxbc_kernel(const float* __restrict__ zx, const float* __restrict__ cw,
                               const float* __restrict__ cb, float* __restrict__ out,
                               int B, int L, int Cc, int d_inner, int d_in_proj)
{
    int t = blockIdx.x * blockDim.x + threadIdx.x;
    int total = B * L * Cc;
    if (t >= total) return;
    int c = t % Cc; int bl = t / Cc;
    int l = bl % L; int b = bl / L;
    float acc = cb[c];
    for (int k = 0; k < 4; ++k) {
        int lj = l + k - 3;
        if (lj < 0) continue;
        acc += zx[((long long)(b * L + lj)) * d_in_proj + d_inner + c] * cw[c * 4 + k];
    }
    out[t] = acc * sigmf(acc);
}

// ---------------------------------------------------------------------------
// Sequential SSD scan. One block per (b,h); 256 threads hold the 64x64 state
// (16 f32 each). y[b,l,h,p] = sum_n h_state*C + xs*D
// ---------------------------------------------------------------------------
__global__ void ssd_scan_kernel(const float* __restrict__ xbc, const float* __restrict__ dt,
                                const float* __restrict__ dA, const float* __restrict__ Dp,
                                float* __restrict__ y,
                                int L, int H, int d_inner, int Cc)
{
    const int b = blockIdx.x / H;
    const int h = blockIdx.x % H;
    const int tid = threadIdx.x;
    const int p  = tid >> 2;
    const int n0 = (tid & 3) << 4;
    __shared__ float sB[64], sC[64], sX[64], sXr[64];
    __shared__ float sdA;
    float hs[16];
#pragma unroll
    for (int j = 0; j < 16; ++j) hs[j] = 0.f;
    const float Dh = Dp[h];
    for (int l = 0; l < L; ++l) {
        const long long bl = (long long)b * L + l;
        const float* row = xbc + bl * Cc;
        if (tid < 64) {
            sB[tid] = row[d_inner + tid];
            sC[tid] = row[d_inner + 64 + tid];
        } else if (tid < 128) {
            int pp = tid - 64;
            float xv = row[h * 64 + pp];
            sXr[pp] = xv;
            sX[pp]  = xv * dt[bl * H + h];
        } else if (tid == 128) {
            sdA = dA[bl * H + h];
        }
        __syncthreads();
        const float dAv = sdA;
        const float xp  = sX[p];
        float part = 0.f;
#pragma unroll
        for (int j = 0; j < 16; ++j) {
            hs[j] = hs[j] * dAv + xp * sB[n0 + j];
            part += hs[j] * sC[n0 + j];
        }
        part += __shfl_xor(part, 1);
        part += __shfl_xor(part, 2);
        if ((tid & 3) == 0)
            y[bl * d_inner + h * 64 + p] = part + sXr[p] * Dh;
        __syncthreads();
    }
}

// y = y * silu(z); y = y * rsqrt(mean(y^2)+eps) * norm_w   (in-place, block per (b,l))
__global__ void gate_rms_kernel(float* __restrict__ y, const float* __restrict__ zx,
                                const float* __restrict__ nw, int d_inner, int d_in_proj)
{
    const long long bl = blockIdx.x;
    float* yr = y + bl * d_inner;
    const float* zr = zx + bl * d_in_proj;
    float ss = 0.f;
    for (int i = threadIdx.x; i < d_inner; i += blockDim.x) {
        float z = zr[i];
        float g = yr[i] * (z * sigmf(z));
        yr[i] = g;
        ss += g * g;
    }
    __shared__ float sred[8];
    __shared__ float sinv;
    float v = ss;
    for (int off = 16; off > 0; off >>= 1) v += __shfl_xor(v, off);
    if ((threadIdx.x & 31) == 0) sred[threadIdx.x >> 5] = v;
    __syncthreads();
    if (threadIdx.x == 0) {
        float tot = 0.f;
        for (int i = 0; i < 8; ++i) tot += sred[i];
        sinv = rsqrtf(tot / (float)d_inner + 1e-5f);
    }
    __syncthreads();
    const float inv = sinv;
    for (int i = threadIdx.x; i < d_inner; i += blockDim.x)
        yr[i] = yr[i] * inv * nw[i];
}

// t = BN(mamba_out^T + tokens) per channel; block per channel
__global__ void bn_residual_kernel(const float* __restrict__ mo, const float* __restrict__ tok,
                                   const float* __restrict__ w, const float* __restrict__ bb,
                                   float* __restrict__ out,
                                   int B, int L, int C)
{
    const int c = blockIdx.x;
    const int BL = B * L;
    float s = 0.f, s2 = 0.f;
    for (int i = threadIdx.x; i < BL; i += blockDim.x) {
        int b = i / L, l = i % L;
        float v = mo[(long long)i * C + c] + tok[((long long)b * C + c) * L + l];
        s += v; s2 += v * v;
    }
    __shared__ float r1[8], r2[8];
    __shared__ float smean, sscale;
    for (int off = 16; off > 0; off >>= 1) { s += __shfl_xor(s, off); s2 += __shfl_xor(s2, off); }
    if ((threadIdx.x & 31) == 0) { r1[threadIdx.x >> 5] = s; r2[threadIdx.x >> 5] = s2; }
    __syncthreads();
    if (threadIdx.x == 0) {
        float t1 = 0.f, t2 = 0.f;
        for (int i = 0; i < 8; ++i) { t1 += r1[i]; t2 += r2[i]; }
        float mean = t1 / (float)BL;
        float var  = t2 / (float)BL - mean * mean;
        smean = mean;
        sscale = rsqrtf(var + 1e-5f);
    }
    __syncthreads();
    const float mean = smean, scale = sscale, wc = w[c], bc = bb[c];
    for (int i = threadIdx.x; i < BL; i += blockDim.x) {
        int b = i / L, l = i % L;
        float v = mo[(long long)i * C + c] + tok[((long long)b * C + c) * L + l];
        out[((long long)b * C + c) * L + l] = (v - mean) * scale * wc + bc;
    }
}

// ===========================================================================
// Host side
// ===========================================================================
struct Blk {
    const float *A_log, *D, *W_in, *W_out, *bn_b, *bn_w, *conv_b, *conv_w,
                *dt_bias, *ff_b1, *ff_b2, *ff_w1, *ff_w2, *norm_w;
};

static void fill_block_sorted(Blk& k, void* const* p, int& i) {
    k.A_log  = (const float*)p[i++]; k.D      = (const float*)p[i++];
    k.W_in   = (const float*)p[i++]; k.W_out  = (const float*)p[i++];
    k.bn_b   = (const float*)p[i++]; k.bn_w   = (const float*)p[i++];
    k.conv_b = (const float*)p[i++]; k.conv_w = (const float*)p[i++];
    k.dt_bias= (const float*)p[i++]; k.ff_b1  = (const float*)p[i++];
    k.ff_b2  = (const float*)p[i++]; k.ff_w1  = (const float*)p[i++];
    k.ff_w2  = (const float*)p[i++]; k.norm_w = (const float*)p[i++];
}
static void fill_block_ins(Blk& k, void* const* p, int& i) {
    k.W_in   = (const float*)p[i++]; k.conv_w = (const float*)p[i++];
    k.conv_b = (const float*)p[i++]; k.dt_bias= (const float*)p[i++];
    k.A_log  = (const float*)p[i++]; k.D      = (const float*)p[i++];
    k.norm_w = (const float*)p[i++]; k.W_out  = (const float*)p[i++];
    k.bn_w   = (const float*)p[i++]; k.bn_b   = (const float*)p[i++];
    k.ff_w1  = (const float*)p[i++]; k.ff_b1  = (const float*)p[i++];
    k.ff_w2  = (const float*)p[i++]; k.ff_b2  = (const float*)p[i++];
}

extern "C" void kernel_launch(void* const* d_in, const int* in_sizes, int n_in,
                              void* d_out, int out_size, void* d_ws, size_t ws_size,
                              hipStream_t stream)
{
    const float* image  = (const float*)d_in[0];
    const float* c1w    = (const float*)d_in[1];
    const float* c2w    = (const float*)d_in[2];

    // Parse params: tolerate sorted-key (jax pytree) or insertion-order dict
    // flattening, and conv-before-blocks or blocks-before-conv layer layout.
    Blk blk[2][2];
    const float* lconv[2] = {nullptr, nullptr};
    const int convSize[2] = {512 * 256 * 7, 1024 * 512 * 7};
    int idx = 3;
    for (int li = 0; li < 2; ++li) {
        bool convFirst = (idx < n_in && in_sizes[idx] == convSize[li]);
        if (convFirst) lconv[li] = (const float*)d_in[idx++];
        bool sortedKeys = (idx < n_in && in_sizes[idx] <= 64);  // A_log (H) vs W_in (huge)
        for (int bi = 0; bi < 2; ++bi) {
            if (sortedKeys) fill_block_sorted(blk[li][bi], d_in, idx);
            else            fill_block_ins(blk[li][bi], d_in, idx);
        }
        if (!convFirst) lconv[li] = (const float*)d_in[idx++];
    }

    // Workspace arena (fp32 elements)
    float* w = (float*)d_ws;
    size_t o = 0;
    auto alloc = [&](size_t n) { float* r = w + o; o += n; return r; };
    float* T0  = alloc(1048576);  // token ping
    float* T1  = alloc(1048576);  // token pong
    float* zx  = alloc(2244608);  // zxbcdt (max BL * d_in_proj)
    float* xT  = alloc(524288);   // tokens transposed (BL, C)
    float* dtb = alloc(16384);    // dt (BL,H)
    float* dAb = alloc(16384);    // dA (BL,H)
    float* xbc = alloc(1179648);  // silu(conv(xBC)) (B,L,Cc)
    float* yb  = alloc(1048576);  // scan output / gated / normed (BL, d_inner)
    float* mo  = alloc(524288);   // mamba out (BL, C)
    float* tb  = alloc(524288);   // bn output (B,C,L)
    float* h1  = alloc(1048576);  // FF hidden (B, 2C, L)
    (void)ws_size;

    const int B = 4;

    // Stem convs
    conv2d_s2_kernel<<<(B * 2 * 128 * 1024 + 255) / 256, 256, 0, stream>>>(
        image, c1w, T0, B, 1, 256, 2048, 2, 128, 1024);
    conv2d_s2_kernel<<<(B * 4 * 64 * 512 + 255) / 256, 256, 0, stream>>>(
        T0, c2w, T1, B, 2, 128, 1024, 4, 64, 512);

    float* cur = T1;   // tokens (B, 256, 512) == conv2 output memory layout
    float* alt = T0;

    int ch = 256, L = 512;
    for (int li = 0; li < 2; ++li) {
        const int C = ch * 2;
        const int Lc = L / 2;
        // conv1d_s2: cur (B,ch,L) -> alt (B,C,Lc)
        conv1d_s2_kernel<<<(B * C * Lc + 255) / 256, 256, 0, stream>>>(
            cur, lconv[li], alt, B, ch, L, C, Lc);
        { float* t = cur; cur = alt; alt = t; }

        const int d_inner   = 2 * C;
        const int H         = d_inner / 64;
        const int d_in_proj = 2 * d_inner + 128 + H;
        const int Cc        = d_inner + 128;
        const int F         = 2 * C;
        const int BL        = B * Lc;

        for (int bi = 0; bi < 2; ++bi) {
            const Blk& K = blk[li][bi];
            // 1) tokens^T
            transpose_CL_kernel<<<(B * C * Lc + 255) / 256, 256, 0, stream>>>(cur, xT, B, C, Lc);
            // 2) zxbcdt = xT @ W_in   (M=BL, N=d_in_proj, K=C)
            {
                dim3 g((d_in_proj + 31) / 32, BL / 64, 1);
                wmma_gemm_kernel<<<g, 128, 0, stream>>>(xT, K.W_in, zx, nullptr,
                    BL, d_in_proj, C, C, d_in_proj, d_in_proj, 0, 0, 0, 0);
            }
            // 3) dt, dA
            dtprep_kernel<<<(BL * H + 255) / 256, 256, 0, stream>>>(
                zx, K.dt_bias, K.A_log, dtb, dAb, BL, H, d_in_proj);
            // 4) depthwise conv + SiLU
            convxbc_kernel<<<(BL * Cc + 255) / 256, 256, 0, stream>>>(
                zx, K.conv_w, K.conv_b, xbc, B, Lc, Cc, d_inner, d_in_proj);
            // 5) SSD scan
            ssd_scan_kernel<<<B * H, 256, 0, stream>>>(
                xbc, dtb, dAb, K.D, yb, Lc, H, d_inner, Cc);
            // 6) gate + RMSNorm (in-place on yb, z from zx)
            gate_rms_kernel<<<BL, 256, 0, stream>>>(yb, zx, K.norm_w, d_inner, d_in_proj);
            // 7) mamba out = yb @ W_out  (M=BL, N=C, K=d_inner)
            {
                dim3 g((C + 31) / 32, BL / 64, 1);
                wmma_gemm_kernel<<<g, 128, 0, stream>>>(yb, K.W_out, mo, nullptr,
                    BL, C, d_inner, d_inner, C, C, 0, 0, 0, 0);
            }
            // 8) residual + batchnorm -> tb (B,C,L)
            bn_residual_kernel<<<C, 256, 0, stream>>>(mo, cur, K.bn_w, K.bn_b, tb, B, Lc, C);
            // 9) h1[b] = relu(ff_w1 @ tb[b] + b1)  (M=F, N=Lc, K=C, batched)
            {
                dim3 g((Lc + 31) / 32, F / 64, B);
                wmma_gemm_kernel<<<g, 128, 0, stream>>>(K.ff_w1, tb, h1, K.ff_b1,
                    F, Lc, C, C, Lc, Lc,
                    0, (long long)C * Lc, (long long)F * Lc, 1);
            }
            // 10) tokens = ff_w2 @ h1[b] + b2 -> alt  (M=C, N=Lc, K=F, batched)
            {
                dim3 g((Lc + 31) / 32, C / 64, B);
                wmma_gemm_kernel<<<g, 128, 0, stream>>>(K.ff_w2, h1, alt, K.ff_b2,
                    C, Lc, F, F, Lc, Lc,
                    0, (long long)F * Lc, (long long)C * Lc, 0);
            }
            { float* t = cur; cur = alt; alt = t; }
        }
        ch = C;
        L = Lc;
    }

    hipMemcpyAsync(d_out, cur, (size_t)out_size * sizeof(float),
                   hipMemcpyDeviceToDevice, stream);
}
